// OptionEncoder_Recurrent_53360673685998
// MI455X (gfx1250) — compile-verified, matrix operand order checked by reading, not run on hardware
//
#include <hip/hip_runtime.h>
#include <hip/hip_bf16.h>
#include <stdint.h>

// ---------------------------------------------------------------------------
// Sizes from the reference
// ---------------------------------------------------------------------------
#define B_   256
#define T_   512
#define IN_  256
#define L_   8
#define C_   16
#define H_   512
#define G4   (4 * H_)              // 2048 gate width
#define IN0  (IN_ + L_ * C_)       // 384  (x | ct)
#define K0C  (IN0 + H_)            // 896  fused layer-0 K  ([x|ct|h0])
#define K1C  (2 * H_)              // 1024 fused layer-1 K  ([h0|h1])
#define KLC  (H_)                  // 512  linear K
#define LC_  (L_ * C_)             // 128
#define TAU_ 0.1f

// LDS row strides (ushort units), padded so lane stride mod 64 dwords == 4
// -> 16 lanes x 16B ds loads hit 64 distinct banks.
#define S0 904    // 896 + 8 ;  904*2/4 = 452 dwords, 452 % 64 = 4
#define S1 1032   // 1024 + 8 ; 1032*2/4 = 516 dwords, 516 % 64 = 4

// workspace layout (byte offsets)
#define OFF_W0 0u
#define OFF_W1 (OFF_W0 + (unsigned)(G4 * K0C * 2))          // 3,670,016
#define OFF_WL (OFF_W1 + (unsigned)(G4 * K1C * 2))          // 7,864,320
#define OFF_B0 (OFF_WL + (unsigned)(LC_ * KLC * 2))         // 7,995,392
#define OFF_B1 (OFF_B0 + (unsigned)(G4 * 4))
#define OFF_BL (OFF_B1 + (unsigned)(G4 * 4))

typedef __attribute__((ext_vector_type(16))) __bf16 v16bf;
typedef __attribute__((ext_vector_type(8)))  float  v8f;

// ---------------------------------------------------------------------------
// helpers
// ---------------------------------------------------------------------------
__device__ __forceinline__ unsigned f2bf_u32(float f) {
    unsigned u = __float_as_uint(f);
    u += 0x7FFFu + ((u >> 16) & 1u);          // round to nearest even
    return u >> 16;
}
__device__ __forceinline__ unsigned short f2bf(float f) {
    return (unsigned short)f2bf_u32(f);
}
__device__ __forceinline__ unsigned pack_bf2(float lo, float hi) {
    return f2bf_u32(lo) | (f2bf_u32(hi) << 16);
}

union V16U { uint4 q[2]; v16bf v; };

// Load one WMMA 16-bit fragment for this lane: 8 bf16 at p, 8 bf16 at p+16.
// LDS pointers -> ds_load_b128 x2 ; global pointers -> global_load_b128 x2.
__device__ __forceinline__ v16bf ld_frag(const unsigned short* p) {
    V16U u;
    u.q[0] = *(const uint4*)(p);
    u.q[1] = *(const uint4*)(p + 16);
    return u.v;
}

__device__ __forceinline__ float sigm(float x) {
    return __builtin_amdgcn_rcpf(1.0f + __expf(-x));
}
__device__ __forceinline__ float tanh_fast(float x) {
    float e = __expf(2.0f * x);
    return (e - 1.0f) * __builtin_amdgcn_rcpf(e + 1.0f);
}

__device__ __forceinline__ unsigned hash_u32(unsigned x) {
    x ^= x >> 16; x *= 0x7feb352dU;
    x ^= x >> 15; x *= 0x846ca68bU;
    x ^= x >> 16; return x;
}
// counter-based gumbel draw, deterministic in (t, b, j)
__device__ __forceinline__ float gumbel_noise(int t, int b, int j) {
    unsigned s = hash_u32(((unsigned)t * 0x9E3779B9u) ^
                          hash_u32((unsigned)(b * LC_ + j) ^ 0x42424242u));
    float u = ((s >> 8) + 0.5f) * (1.0f / 16777216.0f);   // (0,1)
    return -__logf(-__logf(u));
}

// ---------------------------------------------------------------------------
// prep: fuse + pack weights to bf16 (K-major), combine biases
// w0[g][k] k<384 -> Wih0, else Whh0 ; w1[g][k] k<512 -> Wih1, else Whh1
// ---------------------------------------------------------------------------
__global__ void prep_kernel(const float* __restrict__ Wih0, const float* __restrict__ Whh0,
                            const float* __restrict__ bih0, const float* __restrict__ bhh0,
                            const float* __restrict__ Wih1, const float* __restrict__ Whh1,
                            const float* __restrict__ bih1, const float* __restrict__ bhh1,
                            const float* __restrict__ Wlin, const float* __restrict__ blin,
                            unsigned short* __restrict__ w0, unsigned short* __restrict__ w1,
                            unsigned short* __restrict__ wl,
                            float* __restrict__ bb0, float* __restrict__ bb1,
                            float* __restrict__ blf)
{
    const long long n0 = (long long)G4 * K0C;
    const long long n1 = (long long)G4 * K1C;
    const long long nl = (long long)LC_ * KLC;
    const long long total = n0 + n1 + nl + G4 + G4 + LC_;
    for (long long i = (long long)blockIdx.x * blockDim.x + threadIdx.x;
         i < total; i += (long long)gridDim.x * blockDim.x) {
        long long j = i;
        if (j < n0) {
            int g = (int)(j / K0C), k = (int)(j % K0C);
            float v = (k < IN0) ? Wih0[(long long)g * IN0 + k]
                                : Whh0[(long long)g * H_ + (k - IN0)];
            w0[j] = f2bf(v);
            continue;
        }
        j -= n0;
        if (j < n1) {
            int g = (int)(j / K1C), k = (int)(j % K1C);
            float v = (k < H_) ? Wih1[(long long)g * H_ + k]
                               : Whh1[(long long)g * H_ + (k - H_)];
            w1[j] = f2bf(v);
            continue;
        }
        j -= n1;
        if (j < nl) { wl[j] = f2bf(Wlin[j]); continue; }
        j -= nl;
        if (j < G4) { bb0[j] = bih0[j] + bhh0[j]; continue; }
        j -= G4;
        if (j < G4) { bb1[j] = bih1[j] + bhh1[j]; continue; }
        j -= G4;
        blf[j] = blin[j];
    }
}

// ---------------------------------------------------------------------------
// main persistent kernel: one workgroup = one 16-row batch tile, all T steps.
// 256 threads = 8 waves -> 2 waves/SIMD32. Wave w owns hidden units
// [64w, 64w+64) of each layer, in 4 sub-passes; each sub-pass computes gate
// tiles {i,f,g,o} for one 16-unit tile with only 4 live f32 accumulators.
// K loops kept nearly rolled (unroll 2) so the live set fits the VGPR file:
// per iteration pair = 4 ds_load_b128 + 16 global_load_b128 + 8 v_wmma.
// ---------------------------------------------------------------------------
__global__ __launch_bounds__(256, 1)
void lstm_gumbel_kernel(const float* __restrict__ states,
                        const unsigned short* __restrict__ w0,
                        const unsigned short* __restrict__ w1,
                        const unsigned short* __restrict__ wl,
                        const float* __restrict__ bb0, const float* __restrict__ bb1,
                        const float* __restrict__ blf,
                        float* __restrict__ out)
{
    __shared__ unsigned short A0[16 * S0];   // [x(256) | ct(128) | h0(512)] bf16
    __shared__ unsigned short A1[16 * S1];   // [h0(512) | h1(512)] bf16

    const int tid  = threadIdx.x;
    const int w    = tid >> 5;       // wave 0..7
    const int lane = tid & 31;
    const int hsel = lane >> 4;      // K sub-chunk select
    const int col  = lane & 15;      // row for A, column for B/C/D
    const int b0   = blockIdx.x * 16;

    // ---- init LDS state: h = 0, ct = one-hot(0) per option ----
    for (int i = tid; i < 16 * S0; i += 256) A0[i] = 0;
    for (int i = tid; i < 16 * S1; i += 256) A1[i] = 0;
    __syncthreads();
    for (int j = tid; j < 16 * LC_; j += 256) {
        int r = j >> 7, jj = j & 127;
        A0[r * S0 + IN_ + jj] = ((jj & 15) == 0) ? (unsigned short)0x3F80 : (unsigned short)0;
    }

    float c0st[4][8], c1st[4][8];
    #pragma unroll
    for (int s = 0; s < 4; ++s)
        #pragma unroll
        for (int k = 0; k < 8; ++k) { c0st[s][k] = 0.f; c1st[s][k] = 0.f; }

    __syncthreads();

    for (int t = 0; t < T_; ++t) {
        // ---- stage x_t (fp32 -> bf16) into A0[:, 0:256), 16 elems/thread ----
        {
            int row = tid >> 4;                 // 0..15
            int cb  = (tid & 15) * 16;          // 0..240
            const float* sp = states + (((long long)(b0 + row) * T_ + t) * IN_ + cb);
            float4 xa = *(const float4*)sp;
            float4 xb = *(const float4*)(sp + 4);
            float4 xc = *(const float4*)(sp + 8);
            float4 xd = *(const float4*)(sp + 12);
            if (t + 1 < T_) __builtin_prefetch(sp + IN_, 0, 1);   // next step's x
            uint4 pa, pb;
            pa.x = pack_bf2(xa.x, xa.y); pa.y = pack_bf2(xa.z, xa.w);
            pa.z = pack_bf2(xb.x, xb.y); pa.w = pack_bf2(xb.z, xb.w);
            pb.x = pack_bf2(xc.x, xc.y); pb.y = pack_bf2(xc.z, xc.w);
            pb.z = pack_bf2(xd.x, xd.y); pb.w = pack_bf2(xd.z, xd.w);
            *(uint4*)&A0[row * S0 + cb]     = pa;
            *(uint4*)&A0[row * S0 + cb + 8] = pb;
        }
        __syncthreads();

        // =========== layer 0: gates = [x|ct|h0] @ W0cat^T + bias ===========
        float h0reg[4][8];
        #pragma unroll
        for (int s = 0; s < 4; ++s) {
            const int ht = 4 * w + s;            // hidden tile 0..31
            v8f acc[4];
            #pragma unroll
            for (int q = 0; q < 4; ++q) acc[q] = (v8f){};
            const unsigned short* ap  = &A0[col * S0 + hsel * 8];
            const unsigned short* bp0 = w0 + (size_t)(16 * ht + col) * K0C + hsel * 8;
            #pragma clang loop unroll_count(2)
            for (int kc = 0; kc < K0C / 32; ++kc) {
                v16bf a = ld_frag(ap + kc * 32);
                acc[0] = __builtin_amdgcn_wmma_f32_16x16x32_bf16(
                    false, a, false, ld_frag(bp0 + kc * 32), (short)0, acc[0], false, false);
                acc[1] = __builtin_amdgcn_wmma_f32_16x16x32_bf16(
                    false, a, false, ld_frag(bp0 + (size_t)512 * K0C + kc * 32), (short)0, acc[1], false, false);
                acc[2] = __builtin_amdgcn_wmma_f32_16x16x32_bf16(
                    false, a, false, ld_frag(bp0 + (size_t)1024 * K0C + kc * 32), (short)0, acc[2], false, false);
                acc[3] = __builtin_amdgcn_wmma_f32_16x16x32_bf16(
                    false, a, false, ld_frag(bp0 + (size_t)1536 * K0C + kc * 32), (short)0, acc[3], false, false);
            }
            const float bi = bb0[       16 * ht + col];
            const float bf = bb0[ 512 + 16 * ht + col];
            const float bg = bb0[1024 + 16 * ht + col];
            const float bo = bb0[1536 + 16 * ht + col];
            #pragma unroll
            for (int k = 0; k < 8; ++k) {
                float i_ = sigm(acc[0][k] + bi);
                float f_ = sigm(acc[1][k] + bf);
                float g_ = tanh_fast(acc[2][k] + bg);
                float o_ = sigm(acc[3][k] + bo);
                float c  = f_ * c0st[s][k] + i_ * g_;
                c0st[s][k] = c;
                h0reg[s][k] = o_ * tanh_fast(c);
            }
        }
        __syncthreads();     // everyone done reading A0/A1 before h writes
        #pragma unroll
        for (int s = 0; s < 4; ++s)
            #pragma unroll
            for (int k = 0; k < 8; ++k) {
                int row = k + 8 * hsel;
                int cn  = 16 * (4 * w + s) + col;
                unsigned short hb = f2bf(h0reg[s][k]);
                A0[row * S0 + IN0 + cn] = hb;   // h0_prev for next step
                A1[row * S1 + cn]       = hb;   // layer-1 input now
            }
        __syncthreads();

        // =========== layer 1: gates = [h0|h1] @ W1cat^T + bias ===========
        float h1reg[4][8];
        #pragma unroll
        for (int s = 0; s < 4; ++s) {
            const int ht = 4 * w + s;
            v8f acc[4];
            #pragma unroll
            for (int q = 0; q < 4; ++q) acc[q] = (v8f){};
            const unsigned short* ap  = &A1[col * S1 + hsel * 8];
            const unsigned short* bp0 = w1 + (size_t)(16 * ht + col) * K1C + hsel * 8;
            #pragma clang loop unroll_count(2)
            for (int kc = 0; kc < K1C / 32; ++kc) {
                v16bf a = ld_frag(ap + kc * 32);
                acc[0] = __builtin_amdgcn_wmma_f32_16x16x32_bf16(
                    false, a, false, ld_frag(bp0 + kc * 32), (short)0, acc[0], false, false);
                acc[1] = __builtin_amdgcn_wmma_f32_16x16x32_bf16(
                    false, a, false, ld_frag(bp0 + (size_t)512 * K1C + kc * 32), (short)0, acc[1], false, false);
                acc[2] = __builtin_amdgcn_wmma_f32_16x16x32_bf16(
                    false, a, false, ld_frag(bp0 + (size_t)1024 * K1C + kc * 32), (short)0, acc[2], false, false);
                acc[3] = __builtin_amdgcn_wmma_f32_16x16x32_bf16(
                    false, a, false, ld_frag(bp0 + (size_t)1536 * K1C + kc * 32), (short)0, acc[3], false, false);
            }
            const float bi = bb1[       16 * ht + col];
            const float bf = bb1[ 512 + 16 * ht + col];
            const float bg = bb1[1024 + 16 * ht + col];
            const float bo = bb1[1536 + 16 * ht + col];
            #pragma unroll
            for (int k = 0; k < 8; ++k) {
                float i_ = sigm(acc[0][k] + bi);
                float f_ = sigm(acc[1][k] + bf);
                float g_ = tanh_fast(acc[2][k] + bg);
                float o_ = sigm(acc[3][k] + bo);
                float c  = f_ * c1st[s][k] + i_ * g_;
                c1st[s][k] = c;
                h1reg[s][k] = o_ * tanh_fast(c);
            }
        }
        __syncthreads();
        #pragma unroll
        for (int s = 0; s < 4; ++s)
            #pragma unroll
            for (int k = 0; k < 8; ++k) {
                int row = k + 8 * hsel;
                int cn  = 16 * (4 * w + s) + col;
                A1[row * S1 + H_ + cn] = f2bf(h1reg[s][k]);
            }
        __syncthreads();

        // ===== linear + gumbel-softmax: wave w computes option tile l = w =====
        {
            v8f q = (v8f){};
            const unsigned short* ap  = &A1[col * S1 + H_ + hsel * 8];
            const unsigned short* bp0 = wl + (size_t)(w * 16 + col) * KLC + hsel * 8;
            #pragma clang loop unroll_count(2)
            for (int kc = 0; kc < KLC / 32; ++kc) {
                v16bf a = ld_frag(ap + kc * 32);
                q = __builtin_amdgcn_wmma_f32_16x16x32_bf16(
                    false, a, false, ld_frag(bp0 + kc * 32), (short)0, q, false, false);
            }
            const float bq = blf[w * 16 + col];
            // softmax over the 16 classes (N across lanes, within each 16-lane half)
            #pragma unroll
            for (int k = 0; k < 8; ++k) {
                int row = k + 8 * hsel;
                float z = q[k] + bq + gumbel_noise(t, b0 + row, w * 16 + col);
                float m = z;
                m = fmaxf(m, __shfl_xor(m, 1));
                m = fmaxf(m, __shfl_xor(m, 2));
                m = fmaxf(m, __shfl_xor(m, 4));
                m = fmaxf(m, __shfl_xor(m, 8));
                float e = __expf((z - m) * (1.0f / TAU_));
                float s = e;
                s += __shfl_xor(s, 1);
                s += __shfl_xor(s, 2);
                s += __shfl_xor(s, 4);
                s += __shfl_xor(s, 8);
                float y = e * __builtin_amdgcn_rcpf(s);
                out[(((long long)(b0 + row)) * T_ + t) * LC_ + w * 16 + col] = y;
                A0[row * S0 + IN_ + w * 16 + col] = f2bf(y);   // ct for next step
            }
        }
        __syncthreads();
    }
}

// ---------------------------------------------------------------------------
extern "C" void kernel_launch(void* const* d_in, const int* in_sizes, int n_in,
                              void* d_out, int out_size, void* d_ws, size_t ws_size,
                              hipStream_t stream) {
    const float* states = (const float*)d_in[0];
    const float* Wih0   = (const float*)d_in[1];
    const float* Whh0   = (const float*)d_in[2];
    const float* bih0   = (const float*)d_in[3];
    const float* bhh0   = (const float*)d_in[4];
    const float* Wih1   = (const float*)d_in[5];
    const float* Whh1   = (const float*)d_in[6];
    const float* bih1   = (const float*)d_in[7];
    const float* bhh1   = (const float*)d_in[8];
    const float* Wlin   = (const float*)d_in[9];
    const float* blin   = (const float*)d_in[10];

    char* ws = (char*)d_ws;
    unsigned short* w0 = (unsigned short*)(ws + OFF_W0);
    unsigned short* w1 = (unsigned short*)(ws + OFF_W1);
    unsigned short* wl = (unsigned short*)(ws + OFF_WL);
    float* bb0 = (float*)(ws + OFF_B0);
    float* bb1 = (float*)(ws + OFF_B1);
    float* blf = (float*)(ws + OFF_BL);

    long long total = (long long)G4 * K0C + (long long)G4 * K1C +
                      (long long)LC_ * KLC + 2 * G4 + LC_;
    int blocks = (int)((total + 255) / 256);
    prep_kernel<<<blocks, 256, 0, stream>>>(Wih0, Whh0, bih0, bhh0,
                                            Wih1, Whh1, bih1, bhh1,
                                            Wlin, blin,
                                            w0, w1, wl, bb0, bb1, blf);

    lstm_gumbel_kernel<<<B_ / 16, 256, 0, stream>>>(states, w0, w1, wl,
                                                    bb0, bb1, blf,
                                                    (float*)d_out);
}
